// RNN_LSTM_truncated_764504178977
// MI455X (gfx1250) — matrix operand order repaired and checked by reading, not compiled, not run on hardware
//
#include <hip/hip_runtime.h>
#include <hip/hip_bf16.h>

// ---------------------------------------------------------------------------
// LSTM forward (B=64, T=512, I=256, H=512, C=128) for gfx1250 (CDNA5).
//  K0: convert x -> bf16; pack W_ih^T / W_hh^T / W_fc^T into WMMA B-fragment
//      layout (bf16); bias = b_ih + b_hh; zero h/c state.
//  K1: Xg[B*T, 4H] = xb @ WihT + bias      (massive parallel bf16 WMMA GEMM)
//  K2 (x513): per-timestep kernel: stage h_{t-1} into LDS via async copy,
//      fused FC for t-1 + gates GEMM for t (A-fragments from LDS) +
//      elementwise cell update. 16 WGs, each owns a 32-wide hidden slice.
// ---------------------------------------------------------------------------

#define BB   64
#define TT   512
#define II   256
#define HH   512
#define CC   128
#define G4H  2048            // 4*H
#define BT   32768           // B*T

typedef __attribute__((ext_vector_type(16))) __bf16 v16bf;
typedef __attribute__((ext_vector_type(8)))  __bf16 v8bf;
typedef __attribute__((ext_vector_type(8)))  float  v8f;
typedef __attribute__((ext_vector_type(4)))  int    v4i;

typedef __attribute__((address_space(1))) v4i* gv4i_p;   // global-AS vector ptr
typedef __attribute__((address_space(3))) v4i* lv4i_p;   // LDS-AS vector ptr

union V16U { v16bf v; v8bf h[2]; };

#if defined(__has_builtin)
#if __has_builtin(__builtin_amdgcn_global_load_async_to_lds_b128)
#define HAVE_ASYNC_LDS 1
#endif
#if __has_builtin(__builtin_amdgcn_s_wait_asynccnt)
#define HAVE_WAIT_ASYNC 1
#endif
#endif

// Workspace layout (byte offsets; all >= 32B aligned)
#define XB_OFF    ((size_t)0)                       // BT*II bf16    = 16 MB
#define WIH_OFF   ((size_t)16777216)                // 2048*256 bf16 =  1 MB
#define WHH_OFF   ((size_t)17825792)                // 2048*512 bf16 =  2 MB
#define WFC_OFF   ((size_t)19922944)                //  128*512 bf16 = 128 KB
#define BIAS_OFF  ((size_t)20054016)                // 2048 f32
#define HB0_OFF   ((size_t)20062208)                // 64*512 bf16
#define HB1_OFF   ((size_t)20127744)                // 64*512 bf16
#define CST_OFF   ((size_t)20193280)                // 64*512 f32
#define XG_OFF    ((size_t)20971520)                // BT*4H f32 = 256 MB

// -------------------------- K0: conversions --------------------------------

__global__ void k_cvt_x(const float* __restrict__ x, __bf16* __restrict__ xb, int n) {
  int i = blockIdx.x * blockDim.x + threadIdx.x;
  int stride = gridDim.x * blockDim.x;
  for (; i < n; i += stride) xb[i] = (__bf16)x[i];
}

// Pack W[N][K] (row major, fp32) into bf16 WMMA B-fragment layout:
// for n-tile nt and 32-deep K chunk kk, lane l holds 16 bf16 of column
// n = nt*16 + (l&15): slots 0-7 = K 8h..8h+7, slots 8-15 = K 16+8h..16+8h+7 (h = l>>4).
__global__ void k_pack_b(const float* __restrict__ W, __bf16* __restrict__ dst,
                         int N, int K) {
  int idx = blockIdx.x * blockDim.x + threadIdx.x;
  int total = N * K;
  if (idx >= total) return;
  int n = idx / K, k = idx - n * K;
  int ntile = n >> 4, nl = n & 15;
  int kk = k >> 5, kr = k & 31;
  int lane = (((kr >> 3) & 1) << 4) | nl;
  int slot = (kr & 7) | ((kr >> 4) << 3);
  int kkmax = K >> 5;
  size_t o = (((size_t)(ntile * kkmax + kk) * 32) + lane) * 16 + slot;
  dst[o] = (__bf16)W[idx];
}

__global__ void k_init(const float* __restrict__ b_ih, const float* __restrict__ b_hh,
                       float* __restrict__ bias, __bf16* __restrict__ hb0,
                       __bf16* __restrict__ hb1, float* __restrict__ cst) {
  int i = blockIdx.x * blockDim.x + threadIdx.x;
  if (i < G4H) bias[i] = b_ih[i] + b_hh[i];
  if (i < BB * HH) {
    hb0[i] = (__bf16)0.0f;
    hb1[i] = (__bf16)0.0f;
    cst[i] = 0.0f;
  }
}

// -------------------------- K1: Xg = xb @ WihT + bias ----------------------
// 2048 M-tiles x 128 N-tiles; each wave computes 16x64 (4 N-tiles), K=256.
__global__ void k_xgemm(const __bf16* __restrict__ xb,
                        const __bf16* __restrict__ wih,
                        const float* __restrict__ bias,
                        float* __restrict__ Xg) {
  const int lane = threadIdx.x & 31;
  const int wave = threadIdx.x >> 5;
  const int job  = blockIdx.x * 8 + wave;     // 65536 wave jobs
  const int ng   = job & 31;                  // group of 4 N-tiles
  const int mt   = job >> 5;                  // M-tile 0..2047
  const int nl = lane & 15, hi = lane >> 4;

  const __bf16* arow = xb + (size_t)(mt * 16 + nl) * II;
  v8f acc[4] = {};

  for (int kk = 0; kk < 8; ++kk) {
    const int k0 = kk * 32;
    V16U a;
    a.h[0] = *(const v8bf*)(arow + k0 + hi * 8);
    a.h[1] = *(const v8bf*)(arow + k0 + 16 + hi * 8);
#pragma unroll
    for (int j = 0; j < 4; ++j) {
      const int nt = ng * 4 + j;
      const __bf16* bp = wih + (((size_t)(nt * 8 + kk) * 32) + lane) * 16;
      if (kk < 7) __builtin_prefetch(bp + 32 * 16, 0, 1);
      v16bf b = *(const v16bf*)bp;
      acc[j] = __builtin_amdgcn_wmma_f32_16x16x32_bf16(
          false, a.v, false, b, (short)0, acc[j], false, false);
    }
  }
#pragma unroll
  for (int j = 0; j < 4; ++j) {
    const int col = (ng * 4 + j) * 16 + nl;
    const float bv = bias[col];
#pragma unroll
    for (int r = 0; r < 8; ++r) {
      const int row = mt * 16 + hi * 8 + r;
      Xg[(size_t)row * G4H + col] = acc[j][r] + bv;
    }
  }
}

// -------------------------- K2: per-timestep kernel ------------------------
// grid = 16 WGs x 256 threads. WG g owns hidden slice [g*32, g*32+32).
//  Stage:   h_{t-1} (64 KB bf16) -> LDS (async global->LDS copy)
//  Phase 1 (t>0):  out[:, t-1, :] = h_{t-1} @ WfcT + b_fc   (waves 0..1)
//  Phase 2 (t<T):  gates = Xg[:,t,:] + h_{t-1} @ WhhT ; LDS combine; c/h update
// The 32 KB gate-exchange buffer aliases the 64 KB h staging buffer (all h
// reads complete before gate writes; enforced by a barrier).
__global__ void k_step(int t,
                       const __bf16* __restrict__ hbR, __bf16* __restrict__ hbW,
                       float* __restrict__ cst,
                       const float* __restrict__ Xg,
                       const __bf16* __restrict__ whh,
                       const __bf16* __restrict__ wfc,
                       const float* __restrict__ bfc,
                       float* __restrict__ out) {
  __shared__ __align__(16) char smem_raw[BB * HH * 2];   // 64 KB
  __bf16* shb = (__bf16*)smem_raw;                       // staged h_{t-1}
  float (*gl)[BB][32] = (float (*)[BB][32])smem_raw;     // gates [4][64][32]

  const int lane = threadIdx.x & 31;
  const int wave = threadIdx.x >> 5;
  const int g = blockIdx.x;                 // 0..15
  const int nl = lane & 15, hi = lane >> 4;

  // ---- Stage h_{t-1} into LDS: 4096 x 16B chunks, 16 per thread ----
#pragma unroll
  for (int i = 0; i < 16; ++i) {
    const int chunk = threadIdx.x + i * 256;   // 0..4095
#if defined(HAVE_ASYNC_LDS)
    __builtin_amdgcn_global_load_async_to_lds_b128(
        (gv4i_p)(hbR + chunk * 8),
        (lv4i_p)(shb + chunk * 8),
        0, 0);
#else
    ((v4i*)shb)[chunk] = ((const v4i*)hbR)[chunk];
#endif
  }
#if defined(HAVE_ASYNC_LDS)
#if defined(HAVE_WAIT_ASYNC)
  __builtin_amdgcn_s_wait_asynccnt(0);
#else
  asm volatile("s_wait_asynccnt 0x0" ::: "memory");
#endif
#endif
  __syncthreads();

  // ---- Phase 1: FC for previous timestep (A-fragments from LDS) ----
  if (t > 0 && wave < 2) {
    const int tile = g * 2 + wave;          // 0..31 = 4 Mtiles x 8 Ntiles
    const int mt = tile >> 3, nt = tile & 7;
    const __bf16* arow = shb + (size_t)(mt * 16 + nl) * HH;
    v8f acc = {};
    for (int kk = 0; kk < 16; ++kk) {
      const int k0 = kk * 32;
      V16U a;
      a.h[0] = *(const v8bf*)(arow + k0 + hi * 8);
      a.h[1] = *(const v8bf*)(arow + k0 + 16 + hi * 8);
      v16bf b = *(const v16bf*)(wfc + (((size_t)(nt * 16 + kk) * 32) + lane) * 16);
      acc = __builtin_amdgcn_wmma_f32_16x16x32_bf16(
          false, a.v, false, b, (short)0, acc, false, false);
    }
    const int col = nt * 16 + nl;
    const float bv = bfc[col];
#pragma unroll
    for (int r = 0; r < 8; ++r) {
      const int b_ = mt * 16 + hi * 8 + r;
      out[(size_t)b_ * (TT * CC) + (size_t)(t - 1) * CC + col] = acc[r] + bv;
    }
  }

  // ---- Phase 2: recurrent gates GEMM (A-fragments from LDS) ----
  v8f acc[4] = {};
  int nt[4];
  if (t < TT) {
    const int mt = wave >> 1;               // batch tile 0..3
    const int gp = (wave & 1) * 2;          // gates {gp, gp+1}
#pragma unroll
    for (int e = 0; e < 4; ++e) {
      const int q = gp + (e >> 1), j = e & 1;
      nt[e] = q * 32 + g * 2 + j;           // N-tile in [0,128)
    }
    const __bf16* arow = shb + (size_t)(mt * 16 + nl) * HH;
    for (int kk = 0; kk < 16; ++kk) {
      const int k0 = kk * 32;
      V16U a;
      a.h[0] = *(const v8bf*)(arow + k0 + hi * 8);
      a.h[1] = *(const v8bf*)(arow + k0 + 16 + hi * 8);
#pragma unroll
      for (int e = 0; e < 4; ++e) {
        v16bf b = *(const v16bf*)(whh + (((size_t)(nt[e] * 16 + kk) * 32) + lane) * 16);
        acc[e] = __builtin_amdgcn_wmma_f32_16x16x32_bf16(
            false, a.v, false, b, (short)0, acc[e], false, false);
      }
    }
  }
  __syncthreads();   // all reads of shb complete; LDS now reused for gates

  if (t < TT) {
    const int mt = wave >> 1;
    const int gp = (wave & 1) * 2;
    // add precomputed x-projection (already includes bias), stash in LDS
#pragma unroll
    for (int e = 0; e < 4; ++e) {
      const int q = gp + (e >> 1), j = e & 1;
      const int colg = nt[e] * 16 + nl;     // column in [0, 4H)
#pragma unroll
      for (int r = 0; r < 8; ++r) {
        const int b_ = mt * 16 + hi * 8 + r;
        const float v = acc[e][r] + Xg[((size_t)b_ * TT + t) * G4H + colg];
        gl[q][b_][j * 16 + nl] = v;
      }
    }
  }
  __syncthreads();

  // ---- Phase 2b: elementwise cell update for this WG's hidden slice ----
  if (t < TT) {
#pragma unroll
    for (int u = 0; u < 8; ++u) {
      const int idx = threadIdx.x * 8 + u;  // 0..2047 = 64 batch x 32 hidden
      const int b_ = idx >> 5, hl = idx & 31;
      const int h = g * 32 + hl;
      float ig = gl[0][b_][hl];
      float fg = gl[1][b_][hl];
      float gg = gl[2][b_][hl];
      float og = gl[3][b_][hl];
      ig = 1.0f / (1.0f + __expf(-ig));
      fg = 1.0f / (1.0f + __expf(-fg));
      og = 1.0f / (1.0f + __expf(-og));
      gg = tanhf(gg);
      const size_t ci = (size_t)b_ * HH + h;
      const float cn = fg * cst[ci] + ig * gg;
      cst[ci] = cn;
      hbW[ci] = (__bf16)(og * tanhf(cn));
    }
  }
}

// ---------------------------------------------------------------------------

extern "C" void kernel_launch(void* const* d_in, const int* in_sizes, int n_in,
                              void* d_out, int out_size, void* d_ws, size_t ws_size,
                              hipStream_t stream) {
  const float* x    = (const float*)d_in[0];
  const float* W_ih = (const float*)d_in[1];
  const float* W_hh = (const float*)d_in[2];
  const float* b_ih = (const float*)d_in[3];
  const float* b_hh = (const float*)d_in[4];
  const float* W_fc = (const float*)d_in[5];
  const float* b_fc = (const float*)d_in[6];
  float* out = (float*)d_out;

  char* ws = (char*)d_ws;
  __bf16* xb   = (__bf16*)(ws + XB_OFF);
  __bf16* wih  = (__bf16*)(ws + WIH_OFF);
  __bf16* whh  = (__bf16*)(ws + WHH_OFF);
  __bf16* wfc  = (__bf16*)(ws + WFC_OFF);
  float*  bias = (float*)(ws + BIAS_OFF);
  __bf16* hb0  = (__bf16*)(ws + HB0_OFF);
  __bf16* hb1  = (__bf16*)(ws + HB1_OFF);
  float*  cst  = (float*)(ws + CST_OFF);
  float*  Xg   = (float*)(ws + XG_OFF);

  // K0: conversions / packing / state init
  k_cvt_x<<<8192, 256, 0, stream>>>(x, xb, BT * II);
  k_pack_b<<<(G4H * II + 255) / 256, 256, 0, stream>>>(W_ih, wih, G4H, II);
  k_pack_b<<<(G4H * HH + 255) / 256, 256, 0, stream>>>(W_hh, whh, G4H, HH);
  k_pack_b<<<(CC * HH + 255) / 256, 256, 0, stream>>>(W_fc, wfc, CC, HH);
  k_init<<<(BB * HH + 255) / 256, 256, 0, stream>>>(b_ih, b_hh, bias, hb0, hb1, cst);

  // K1: x-projection for all timesteps (parallel, ~1/3 of all FLOPs)
  k_xgemm<<<8192, 256, 0, stream>>>(xb, wih, bias, Xg);

  // K2: serial recurrence; t == TT does only the final FC
  for (int t = 0; t <= TT; ++t) {
    __bf16* hbR = (t & 1) ? hb1 : hb0;
    __bf16* hbW = (t & 1) ? hb0 : hb1;
    k_step<<<16, 256, 0, stream>>>(t, hbR, hbW, cst, Xg, whh, wfc, b_fc, out);
  }
}
